// ClassificationModel_19799799235164
// MI455X (gfx1250) — matrix operand order repaired and checked by reading, not compiled, not run on hardware
//
#include <hip/hip_runtime.h>
#include <math.h>

#define IN_F   1024
#define VOCAB  50257
#define NTOK   8192
#define KSTEPS (IN_F / 32)
#define LDS_STRIDE 40   // 32 K-elems + 8 pad (ushorts) -> conflict-free b128 frag loads

typedef __attribute__((ext_vector_type(16))) __bf16 bf16x16;
typedef __attribute__((ext_vector_type(8)))  float  v8f;
typedef __attribute__((ext_vector_type(4)))  float  f4;   // plain vector: AS(1)-loadable

union Frag {
    bf16x16 v;
    uint4   q[2];
};

// packed fp32x2 -> bf16x2 in one VALU op (D[15:0]=cvt(lo), D[31:16]=cvt(hi))
__device__ __forceinline__ unsigned int pk_bf16(float lo, float hi) {
    unsigned int d;
    asm("v_cvt_pk_bf16_f32 %0, %1, %2" : "=v"(d) : "v"(lo), "v"(hi));
    return d;
}

// Force AS(1) so the compiler emits global_load_b128, not flat_load_b128.
// (FLAT increments DScnt too, which would make the lds_barrier's
//  s_wait_dscnt 0 stall on in-flight next-tile loads.)
__device__ __forceinline__ f4 gload4(const float* p) {
    return *(const __attribute__((address_space(1))) f4*)(unsigned long long)p;
}

// Workgroup barrier with LDS-only visibility: waits DScnt==0 (tile stores + frag
// loads landed) then signal/wait. Unlike __syncthreads(), does NOT wait loadcnt==0,
// so in-flight global loads for the next tile keep flying through the barrier.
__device__ __forceinline__ void lds_barrier() {
    asm volatile("s_wait_dscnt 0x0\n\t"
                 "s_barrier_signal -1\n\t"
                 "s_barrier_wait -1" ::: "memory");
}

__global__ __launch_bounds__(256)
void gemm_bf16_wmma(const float* __restrict__ A,   // [NTOK, IN_F]
                    const float* __restrict__ W,   // [VOCAB, IN_F]
                    float* __restrict__ logits) {  // [NTOK, VOCAB]
    __shared__ __align__(16) unsigned short sA[2][128 * LDS_STRIDE];
    __shared__ __align__(16) unsigned short sB[2][128 * LDS_STRIDE];

    const int tid  = threadIdx.x;
    const int lane = tid & 31;
    const int wave = tid >> 5;
    const int wm   = wave >> 2;   // 0..1 : wave M position
    const int wn   = wave & 3;    // 0..3 : wave N position
    const int h    = lane >> 4;   // half-wave select
    const int ml   = lane & 15;

    const int blockM = blockIdx.x * 128;   // M fast-varying -> W tile stays hot in L2
    const int blockN = blockIdx.y * 128;

    // per-thread staging coords + global pointers (vocab clamp folded in: OOB rows
    // feed accumulator lanes that are never stored, so any valid row works)
    const float* pa[4];
    const float* pw[4];
    int stR[4], stC[4];
#pragma unroll
    for (int i = 0; i < 4; ++i) {
        const int slot = i * 256 + tid;     // 1024 float4-slots = 128 rows x 8 col-groups
        const int r = slot >> 3;
        const int c = (slot & 7) * 4;
        stR[i] = r; stC[i] = c;
        pa[i] = A + (size_t)(blockM + r) * IN_F + c;
        int v = blockN + r;
        v = (v < VOCAB) ? v : (VOCAB - 1);
        pw[i] = W + (size_t)v * IN_F + c;
    }

    v8f acc[4][2];
#pragma unroll
    for (int i = 0; i < 4; ++i)
#pragma unroll
        for (int j = 0; j < 2; ++j)
#pragma unroll
            for (int e = 0; e < 8; ++e)
                acc[i][j][e] = 0.0f;

    f4 ra[4], rw[4];
#pragma unroll
    for (int i = 0; i < 4; ++i) {
        ra[i] = gload4(pa[i]);  pa[i] += 32;
        rw[i] = gload4(pw[i]);  pw[i] += 32;
    }

#pragma unroll 2
    for (int kt = 0; kt < KSTEPS; ++kt) {
        const int b = kt & 1;
        // ---- issue next tile's global loads first (independent of staging) ----
        f4 ra2[4], rw2[4];
        if (kt + 1 < KSTEPS) {
#pragma unroll
            for (int i = 0; i < 4; ++i) {
                ra2[i] = gload4(pa[i]);  pa[i] += 32;
                rw2[i] = gload4(pw[i]);  pw[i] += 32;
            }
        }
        // ---- convert staged tile to bf16, store into buffer b ----
#pragma unroll
        for (int i = 0; i < 4; ++i) {
            uint2 qa, qb;
            qa.x = pk_bf16(ra[i].x, ra[i].y);
            qa.y = pk_bf16(ra[i].z, ra[i].w);
            qb.x = pk_bf16(rw[i].x, rw[i].y);
            qb.y = pk_bf16(rw[i].z, rw[i].w);
            *(uint2*)(&sA[b][stR[i] * LDS_STRIDE + stC[i]]) = qa;
            *(uint2*)(&sB[b][stR[i] * LDS_STRIDE + stC[i]]) = qb;
        }
        lds_barrier();     // DS-only fence; next-tile global loads stay in flight

        // ---- fragment loads: contiguous ds_load_b128, exact WMMA VGPR layouts ----
        Frag a[4], bb[2];
#pragma unroll
        for (int mt = 0; mt < 4; ++mt) {
            const int m = wm * 64 + mt * 16 + ml;
            const unsigned short* base = &sA[b][m * LDS_STRIDE];
            a[mt].q[0] = *(const uint4*)(base + h * 8);        // K = h*8 .. +7
            a[mt].q[1] = *(const uint4*)(base + 16 + h * 8);   // K = 16+h*8 .. +7
        }
#pragma unroll
        for (int nt = 0; nt < 2; ++nt) {
            const int n = wn * 32 + nt * 16 + ml;
            const unsigned short* base = &sB[b][n * LDS_STRIDE];
            bb[nt].q[0] = *(const uint4*)(base + h * 8);       // K = h*8 .. +7
            bb[nt].q[1] = *(const uint4*)(base + 16 + h * 8);  // K = 16+h*8 .. +7
        }
#pragma unroll
        for (int mt = 0; mt < 4; ++mt)
#pragma unroll
            for (int nt = 0; nt < 2; ++nt)
                acc[mt][nt] = __builtin_amdgcn_wmma_f32_16x16x32_bf16(
                    false, a[mt].v, false, bb[nt].v,
                    (short)0, acc[mt][nt], false, false);

#pragma unroll
        for (int i = 0; i < 4; ++i) { ra[i] = ra2[i]; rw[i] = rw2[i]; }
    }

    // ---- store C tiles nontemporally (streamed 1.65GB; keep W resident in L2) ----
#pragma unroll
    for (int mt = 0; mt < 4; ++mt) {
#pragma unroll
        for (int nt = 0; nt < 2; ++nt) {
            const int gm = blockM + wm * 64 + mt * 16 + h * 8;
            const int gn = blockN + wn * 32 + nt * 16 + ml;
            if (gn < VOCAB) {
#pragma unroll
                for (int r = 0; r < 8; ++r)
                    __builtin_nontemporal_store(acc[mt][nt][r],
                        logits + (size_t)(gm + r) * VOCAB + gn);
            }
        }
    }
}

// one block per token row: single-pass online softmax -> nll[row]
__global__ __launch_bounds__(256)
void ce_rows(const float* __restrict__ logits,
             const int* __restrict__ targets,
             float* __restrict__ nll) {
    const int row = blockIdx.x;
    const float* lp = logits + (size_t)row * VOCAB;

    float m = -INFINITY, s = 0.0f;
    for (int j = threadIdx.x; j < VOCAB; j += 256) {
        const float x = __builtin_nontemporal_load(lp + j);
        if (x > m) { s = s * __expf(m - x) + 1.0f; m = x; }
        else       { s += __expf(x - m); }
    }
    __shared__ float sm[256], ss[256];
    sm[threadIdx.x] = m; ss[threadIdx.x] = s;
    __syncthreads();
    for (int off = 128; off > 0; off >>= 1) {
        if (threadIdx.x < off) {
            const float m1 = sm[threadIdx.x],       s1 = ss[threadIdx.x];
            const float m2 = sm[threadIdx.x + off], s2 = ss[threadIdx.x + off];
            const float M = fmaxf(m1, m2);
            sm[threadIdx.x] = M;
            ss[threadIdx.x] = s1 * __expf(m1 - M) + s2 * __expf(m2 - M);
        }
        __syncthreads();
    }
    if (threadIdx.x == 0) {
        const int t = targets[row];
        nll[row] = sm[0] + __logf(ss[0]) - lp[t];
    }
}

__global__ __launch_bounds__(256)
void reduce_loss(const float* __restrict__ nll, float* __restrict__ loss) {
    __shared__ float sb[256];
    float s = 0.0f;
    for (int i = threadIdx.x; i < NTOK; i += 256) s += nll[i];
    sb[threadIdx.x] = s;
    __syncthreads();
    for (int off = 128; off > 0; off >>= 1) {
        if (threadIdx.x < off) sb[threadIdx.x] += sb[threadIdx.x + off];
        __syncthreads();
    }
    if (threadIdx.x == 0) *loss = sb[0] / (float)NTOK;
}

extern "C" void kernel_launch(void* const* d_in, const int* in_sizes, int n_in,
                              void* d_out, int out_size, void* d_ws, size_t ws_size,
                              hipStream_t stream) {
    const float* inputs  = (const float*)d_in[0];   // [NTOK, IN_F]
    const int*   targets = (const int*)d_in[1];     // [NTOK]
    const float* Wm      = (const float*)d_in[2];   // [VOCAB, IN_F]

    float* logits = (float*)d_out;                           // [NTOK, VOCAB]
    float* loss   = logits + (size_t)NTOK * VOCAB;           // scalar
    float* nll    = (float*)d_ws;                            // [NTOK] scratch

    dim3 grid(NTOK / 128, (VOCAB + 127) / 128);              // x = M (fast) -> W-tile L2 reuse
    gemm_bf16_wmma<<<grid, 256, 0, stream>>>(inputs, Wm, logits);
    ce_rows<<<NTOK, 256, 0, stream>>>(logits, targets, nll);
    reduce_loss<<<1, 256, 0, stream>>>(nll, loss);
}
